// MoELayer_58420145160626
// MI455X (gfx1250) — matrix operand order repaired and checked by reading, not compiled, not run on hardware
//
#include <hip/hip_runtime.h>
#include <hip/hip_bf16.h>

// ---------------- problem constants (match reference) ----------------
#define NTOK  16384      // B*S = 8*2048
#define DDIM  512
#define HDIM  2048
#define NEXP  8
#define KSEL  2
#define TILE_M 64
#define TILES_PER_E 256  // NTOK / TILE_M
#define HPAD (HDIM + 8)  // LDS row padding: 16 consecutive rows hit distinct bank groups

typedef __attribute__((ext_vector_type(16))) __bf16 v16bf;
typedef __attribute__((ext_vector_type(8)))  __bf16 v8bf;
typedef __attribute__((ext_vector_type(8)))  float  v8f;

// ---------------- routing: per-expert token lists ----------------
__global__ void route_kernel(const int* __restrict__ assign,
                             int* __restrict__ counts,
                             int* __restrict__ lists) {
    int n = blockIdx.x * blockDim.x + threadIdx.x;
    if (n >= NTOK) return;
    int e0 = assign[2 * n];
    int e1 = assign[2 * n + 1];
    int p = atomicAdd(&counts[e0], 1);
    lists[e0 * NTOK + p] = n;
    if (e1 != e0) {
        p = atomicAdd(&counts[e1], 1);
        lists[e1 * NTOK + p] = n;
    }
}

// ---------------- fp32 -> bf16 conversions ----------------
__global__ void cvt_x_kernel(const float* __restrict__ x, __bf16* __restrict__ xb) {
    long i = (long)blockIdx.x * blockDim.x + threadIdx.x;
    xb[i] = (__bf16)x[i];
}

// w layout [E][rows][cols] -> wt layout [E][cols][rows] (bf16, transposed)
__global__ void cvt_wT_kernel(const float* __restrict__ w, __bf16* __restrict__ wt,
                              int rows, int cols) {
    long i = (long)blockIdx.x * blockDim.x + threadIdx.x;
    long per = (long)rows * cols;
    long e = i / per;
    long rem = i - e * per;
    long k = rem / cols;
    long n = rem - k * cols;
    wt[e * per + n * rows + k] = (__bf16)w[i];
}

// ---------------- fused per-tile MoE FFN ----------------
__global__ __launch_bounds__(256, 1)
void moe_kernel(const __bf16* __restrict__ xb,
                const __bf16* __restrict__ w1t,   // [E][H][D] bf16 (transposed)
                const float*  __restrict__ b1,    // [E][H]
                const __bf16* __restrict__ w2t,   // [E][D][H] bf16 (transposed)
                const float*  __restrict__ b2,    // [E][D]
                const int*    __restrict__ counts,
                const int*    __restrict__ lists,
                float*        __restrict__ out) {
    __shared__ __bf16 Hs[TILE_M][HPAD];   // 263168 B < 320 KB WGP LDS
    __shared__ int rowids[TILE_M];

    int e  = blockIdx.x >> 8;             // 256 tiles per expert
    int t  = blockIdx.x & 255;
    int cnt = counts[e];
    int m0 = t * TILE_M;
    if (m0 >= cnt) return;
    int mvalid = min(TILE_M, cnt - m0);

    if (threadIdx.x < TILE_M) {
        int idx = min(m0 + (int)threadIdx.x, cnt - 1);
        rowids[threadIdx.x] = lists[e * NTOK + idx];
    }
    __syncthreads();

    int wave = threadIdx.x >> 5;
    int lane = threadIdx.x & 31;
    int half = lane >> 4;                 // ISA 16x16 layout: lane half
    int r    = lane & 15;
    int rb   = wave >> 1;                 // 4 row-blocks of 16 rows
    int ch   = wave & 1;                  // 2 column halves

    // ================= GEMM1: H = relu(X_sel @ W1 + b1) =================
    // Preload all A fragments for this wave's 16 rows (K = 512 -> 16 frags).
    v16bf afrag[16];
    {
        int row = rowids[rb * 16 + r];
        const __bf16* xrow = xb + (long)row * DDIM;
#pragma unroll
        for (int kk = 0; kk < 16; ++kk) {
            int k0 = kk * 32;
            v8bf p0 = *(const v8bf*)(xrow + k0 + half * 8);
            v8bf p1 = *(const v8bf*)(xrow + k0 + 16 + half * 8);
#pragma unroll
            for (int i = 0; i < 8; ++i) { afrag[kk][i] = p0[i]; afrag[kk][8 + i] = p1[i]; }
        }
    }
    const __bf16* w1e = w1t + (long)e * HDIM * DDIM;
    for (int ct = 0; ct < 64; ++ct) {     // 64 N-tiles of 16 per wave (half of 2048)
        int n0 = ch * 1024 + ct * 16;
        const __bf16* bcol = w1e + (long)(n0 + r) * DDIM + half * 8;
        v8f acc = {};
#pragma unroll
        for (int kk = 0; kk < 16; ++kk) {
            v8bf q0 = *(const v8bf*)(bcol + kk * 32);
            v8bf q1 = *(const v8bf*)(bcol + kk * 32 + 16);
            v16bf bf;
#pragma unroll
            for (int i = 0; i < 8; ++i) { bf[i] = q0[i]; bf[8 + i] = q1[i]; }
            acc = __builtin_amdgcn_wmma_f32_16x16x32_bf16(
                false, afrag[kk], false, bf, (short)0, acc, false, false);
        }
        float bias = b1[e * HDIM + n0 + r];
#pragma unroll
        for (int i = 0; i < 8; ++i) {     // C/D: VGPR i holds row (i + 8*half), col r
            int m = rb * 16 + i + 8 * half;
            float v = acc[i] + bias;
            Hs[m][n0 + r] = (__bf16)(v > 0.0f ? v : 0.0f);
        }
    }
    __syncthreads();

    // ================= GEMM2: Y = H @ W2 + b2 =================
    const __bf16* w2e = w2t + (long)e * DDIM * HDIM;
    v8f acc2[16];
#pragma unroll
    for (int i = 0; i < 16; ++i) acc2[i] = (v8f){};
    for (int kk = 0; kk < 64; ++kk) {     // K = 2048
        int k0 = kk * 32;
        v16bf af;
        {
            const __bf16* hrow = &Hs[rb * 16 + r][0];
            v8bf p0 = *(const v8bf*)(hrow + k0 + half * 8);
            v8bf p1 = *(const v8bf*)(hrow + k0 + 16 + half * 8);
#pragma unroll
            for (int i = 0; i < 8; ++i) { af[i] = p0[i]; af[8 + i] = p1[i]; }
        }
#pragma unroll
        for (int ct = 0; ct < 16; ++ct) { // 16 N-tiles of 16 per wave (half of 512)
            int n0 = ch * 256 + ct * 16;
            const __bf16* bcol = w2e + (long)(n0 + r) * HDIM + k0 + half * 8;
            v8bf q0 = *(const v8bf*)(bcol);
            v8bf q1 = *(const v8bf*)(bcol + 16);
            v16bf bf;
#pragma unroll
            for (int i = 0; i < 8; ++i) { bf[i] = q0[i]; bf[8 + i] = q1[i]; }
            acc2[ct] = __builtin_amdgcn_wmma_f32_16x16x32_bf16(
                false, af, false, bf, (short)0, acc2[ct], false, false);
        }
    }
#pragma unroll
    for (int ct = 0; ct < 16; ++ct) {
        int n = ch * 256 + ct * 16 + r;
        float bias = b2[e * DDIM + n];
#pragma unroll
        for (int i = 0; i < 8; ++i) {
            int m = rb * 16 + i + 8 * half;
            if (m < mvalid) {
                int tok = rowids[m];
                atomicAdd(&out[(long)tok * DDIM + n], 0.5f * (acc2[ct][i] + bias));
            }
        }
    }
}

// ---------------- launcher ----------------
extern "C" void kernel_launch(void* const* d_in, const int* in_sizes, int n_in,
                              void* d_out, int out_size, void* d_ws, size_t ws_size,
                              hipStream_t stream) {
    const float* x      = (const float*)d_in[0];
    const int*   assign = (const int*)d_in[1];   // integer inputs delivered as int32
    const float* W1     = (const float*)d_in[2];
    const float* b1     = (const float*)d_in[3];
    const float* W2     = (const float*)d_in[4];
    const float* b2     = (const float*)d_in[5];
    float* out = (float*)d_out;

    char* ws = (char*)d_ws;
    int* counts = (int*)ws;                                     // 8 ints
    int* lists  = (int*)(ws + 256);                             // E*N ints (512 KB)
    size_t off = 256 + (size_t)NEXP * NTOK * sizeof(int);
    off = (off + 255) & ~(size_t)255;
    __bf16* xbw  = (__bf16*)(ws + off); off += (size_t)NTOK * DDIM * 2;        // 16 MB
    __bf16* w1tw = (__bf16*)(ws + off); off += (size_t)NEXP * DDIM * HDIM * 2; // 16 MB
    __bf16* w2tw = (__bf16*)(ws + off); off += (size_t)NEXP * HDIM * DDIM * 2; // 16 MB

    hipMemsetAsync(counts, 0, NEXP * sizeof(int), stream);
    hipMemsetAsync(out, 0, (size_t)out_size * sizeof(float), stream);

    route_kernel<<<NTOK / 256, 256, 0, stream>>>(assign, counts, lists);
    cvt_x_kernel<<<(NTOK * DDIM) / 256, 256, 0, stream>>>(x, xbw);
    cvt_wT_kernel<<<(NEXP * DDIM * HDIM) / 256, 256, 0, stream>>>(W1, w1tw, DDIM, HDIM);
    cvt_wT_kernel<<<(NEXP * HDIM * DDIM) / 256, 256, 0, stream>>>(W2, w2tw, HDIM, DDIM);

    moe_kernel<<<NEXP * TILES_PER_E, 256, 0, stream>>>(
        xbw, w1tw, b1, w2tw, b2, counts, lists, out);
}
